// Attention_Encoder_46437186404873
// MI455X (gfx1250) — matrix operand order repaired and checked by reading, not compiled, not run on hardware
//
#include <hip/hip_runtime.h>
#include <stdint.h>

typedef __bf16 bf16_t;
typedef bf16_t v16bf __attribute__((ext_vector_type(16)));
typedef bf16_t v8bf  __attribute__((ext_vector_type(8)));
typedef float  v8f   __attribute__((ext_vector_type(8)));

#define DEV __device__ __forceinline__

static constexpr int BB=8, SS=256, HC=150, VTRAIN=32000;
static constexpr int DD=512, HH=8, DKDIM=64, DFF=2048, LLAY=6, NPAIR=32385;

DEV bf16_t f2bf(float f){
  unsigned u = __builtin_bit_cast(unsigned, f);
  unsigned r = u + 0x7FFFu + ((u>>16)&1u);
  unsigned short h = (unsigned short)(r>>16);
  return __builtin_bit_cast(bf16_t, h);
}
DEV float sigm(float x){ return 1.f/(1.f+expf(-x)); }

DEV float blk_sum(float v){
  __shared__ float red[256];
  red[threadIdx.x]=v; __syncthreads();
  for(int s=128;s>0;s>>=1){ if((int)threadIdx.x<s) red[threadIdx.x]+=red[threadIdx.x+s]; __syncthreads(); }
  float r=red[0]; __syncthreads(); return r;
}

// ---------------------------------------------------------------- WMMA GEMM
// C[M,N] = A[M,K] (bf16, row-major) @ Bt[N,K]^T (bf16, row-major N x K)
// Wave32 fragment layouts used here (A/B in contiguous k order, not the ISA table order):
//   A 16x32: lane row = lane%16; element e -> K = 16*(lane/16) + e
//   B 32x16 (from Bt): lane col = lane%16; element e -> K = 16*(lane/16) + e
//   C 16x16 f32: vgpr j -> row = j + 8*(lane/16), col = lane%16
// Block = 4 waves, block tile 128(M) x 64(N): each wave owns 2 M-subtiles x 4 N-subtiles
// (8 independent WMMAs per k-step). Software pipeline: steady-state loop has
// unconditional next-step loads; last two k-steps peeled (K is always a
// multiple of 64 here, so the step count is even).
struct GemmP {
  const bf16_t* A; long long Aso, Asi; int lda;
  const bf16_t* Bt; long long Bso, Bsi; int ldb;
  float* Cf; bf16_t* Cb; long long Cso, Csi; int ldc;
  int K, Hdiv; const float* bias; int relu, mode; // mode 0=f32, 1=bf16, 2=bf16 V^T scatter
};

DEV v16bf load_a(const bf16_t* A, int lda, int row, int k0, int hi){
  union { v16bf v; v8bf h[2]; } f;
  const bf16_t* p = A + (long long)row*lda + k0 + hi*16;
  f.h[0] = *(const v8bf*)(p);
  f.h[1] = *(const v8bf*)(p + 8);
  return f.v;
}
DEV v16bf load_b(const bf16_t* Bt, int ldb, int col, int k0, int hi){
  return *(const v16bf*)(Bt + (long long)col*ldb + k0 + hi*16);
}
DEV v8f wmma_bf16(v16bf a, v16bf b, v8f c){
  return __builtin_amdgcn_wmma_f32_16x16x32_bf16(false, a, false, b, (short)0, c, false, false);
}

__global__ __launch_bounds__(128) void k_gemm(GemmP p){
  int z = blockIdx.z, zo = z / p.Hdiv, zi = z % p.Hdiv;
  const bf16_t* A  = p.A  + zo*p.Aso + zi*p.Asi;
  const bf16_t* Bt = p.Bt + zo*p.Bso + zi*p.Bsi;
  int lane = threadIdx.x & 31, wave = threadIdx.x >> 5;
  int hl = lane & 15, hi = lane >> 4;
  int tm0 = blockIdx.y*128 + wave*16;
  int tm1 = tm0 + 64;
  int tn  = blockIdx.x*64;
  v8f acc[8] = {};

  // double-buffered fragments (static names -> stay in VGPRs)
  v16bf A0a, A1a, B0a, B1a, B2a, B3a;
  v16bf A0b, A1b, B0b, B1b, B2b, B3b;

  A0a = load_a(A, p.lda, tm0 + hl, 0, hi);
  A1a = load_a(A, p.lda, tm1 + hl, 0, hi);
  B0a = load_b(Bt, p.ldb, tn +  0 + hl, 0, hi);
  B1a = load_b(Bt, p.ldb, tn + 16 + hl, 0, hi);
  B2a = load_b(Bt, p.ldb, tn + 32 + hl, 0, hi);
  B3a = load_b(Bt, p.ldb, tn + 48 + hl, 0, hi);

  int kSteps = p.K >> 5;   // always even (K % 64 == 0)
  int k0 = 0;
  for (int s = 0; s < kSteps - 2; s += 2){
    // load 'b' set for k0+32 (unconditional)
    A0b = load_a(A, p.lda, tm0 + hl, k0+32, hi);
    A1b = load_a(A, p.lda, tm1 + hl, k0+32, hi);
    B0b = load_b(Bt, p.ldb, tn +  0 + hl, k0+32, hi);
    B1b = load_b(Bt, p.ldb, tn + 16 + hl, k0+32, hi);
    B2b = load_b(Bt, p.ldb, tn + 32 + hl, k0+32, hi);
    B3b = load_b(Bt, p.ldb, tn + 48 + hl, k0+32, hi);
    __builtin_prefetch((const void*)(A + (long long)(tm0+hl)*p.lda + k0 + 96), 0, 1);
    acc[0] = wmma_bf16(A0a, B0a, acc[0]);
    acc[1] = wmma_bf16(A0a, B1a, acc[1]);
    acc[2] = wmma_bf16(A0a, B2a, acc[2]);
    acc[3] = wmma_bf16(A0a, B3a, acc[3]);
    acc[4] = wmma_bf16(A1a, B0a, acc[4]);
    acc[5] = wmma_bf16(A1a, B1a, acc[5]);
    acc[6] = wmma_bf16(A1a, B2a, acc[6]);
    acc[7] = wmma_bf16(A1a, B3a, acc[7]);
    // load 'a' set for k0+64 (unconditional)
    A0a = load_a(A, p.lda, tm0 + hl, k0+64, hi);
    A1a = load_a(A, p.lda, tm1 + hl, k0+64, hi);
    B0a = load_b(Bt, p.ldb, tn +  0 + hl, k0+64, hi);
    B1a = load_b(Bt, p.ldb, tn + 16 + hl, k0+64, hi);
    B2a = load_b(Bt, p.ldb, tn + 32 + hl, k0+64, hi);
    B3a = load_b(Bt, p.ldb, tn + 48 + hl, k0+64, hi);
    __builtin_prefetch((const void*)(A + (long long)(tm0+hl)*p.lda + k0 + 128), 0, 1);
    acc[0] = wmma_bf16(A0b, B0b, acc[0]);
    acc[1] = wmma_bf16(A0b, B1b, acc[1]);
    acc[2] = wmma_bf16(A0b, B2b, acc[2]);
    acc[3] = wmma_bf16(A0b, B3b, acc[3]);
    acc[4] = wmma_bf16(A1b, B0b, acc[4]);
    acc[5] = wmma_bf16(A1b, B1b, acc[5]);
    acc[6] = wmma_bf16(A1b, B2b, acc[6]);
    acc[7] = wmma_bf16(A1b, B3b, acc[7]);
    k0 += 64;
  }
  // tail: two remaining k-steps (k0 and k0+32)
  A0b = load_a(A, p.lda, tm0 + hl, k0+32, hi);
  A1b = load_a(A, p.lda, tm1 + hl, k0+32, hi);
  B0b = load_b(Bt, p.ldb, tn +  0 + hl, k0+32, hi);
  B1b = load_b(Bt, p.ldb, tn + 16 + hl, k0+32, hi);
  B2b = load_b(Bt, p.ldb, tn + 32 + hl, k0+32, hi);
  B3b = load_b(Bt, p.ldb, tn + 48 + hl, k0+32, hi);
  acc[0] = wmma_bf16(A0a, B0a, acc[0]);
  acc[1] = wmma_bf16(A0a, B1a, acc[1]);
  acc[2] = wmma_bf16(A0a, B2a, acc[2]);
  acc[3] = wmma_bf16(A0a, B3a, acc[3]);
  acc[4] = wmma_bf16(A1a, B0a, acc[4]);
  acc[5] = wmma_bf16(A1a, B1a, acc[5]);
  acc[6] = wmma_bf16(A1a, B2a, acc[6]);
  acc[7] = wmma_bf16(A1a, B3a, acc[7]);
  acc[0] = wmma_bf16(A0b, B0b, acc[0]);
  acc[1] = wmma_bf16(A0b, B1b, acc[1]);
  acc[2] = wmma_bf16(A0b, B2b, acc[2]);
  acc[3] = wmma_bf16(A0b, B3b, acc[3]);
  acc[4] = wmma_bf16(A1b, B0b, acc[4]);
  acc[5] = wmma_bf16(A1b, B1b, acc[5]);
  acc[6] = wmma_bf16(A1b, B2b, acc[6]);
  acc[7] = wmma_bf16(A1b, B3b, acc[7]);

  long long co = zo*p.Cso + zi*p.Csi;
#pragma unroll
  for (int mt = 0; mt < 2; mt++){
    int tm = mt ? tm1 : tm0;
#pragma unroll
    for (int nt = 0; nt < 4; nt++){
#pragma unroll
      for (int j = 0; j < 8; j++){
        int r = tm + hi*8 + j, c = tn + nt*16 + hl;
        float v = acc[mt*4 + nt][j];
        if (p.bias) v += p.bias[c];
        if (p.relu) v = fmaxf(v, 0.f);
        if (p.mode == 0)      p.Cf[co + (long long)r*p.ldc + c] = v;
        else if (p.mode == 1) p.Cb[co + (long long)r*p.ldc + c] = f2bf(v);
        else { // V^T scatter: (b,s)x(h,d) -> Vt[b,h,d,s]  (B=8,S=256,H=8,DK=64)
          int b = r >> 8, s = r & 255, h = c >> 6, d = c & 63;
          p.Cb[((((long long)(b*8 + h))*64 + d) << 8) | s] = f2bf(v);
        }
      }
    }
  }
}

// ------------------------------------------------- weight transpose+convert
__global__ __launch_bounds__(256) void k_transpose(const float* src, bf16_t* dst, int K, int N){
  long long i = (long long)blockIdx.x*256 + threadIdx.x;
  if (i < (long long)K*N){
    int n = (int)(i / K), k = (int)(i % K);
    dst[i] = f2bf(src[(long long)k*N + n]);
  }
}

// ------------------------------------------------------------- char BiLSTM
__global__ __launch_bounds__(160) void k_lstm(const int* char_idxs, const float* char_emb,
    const float* Wih_f, const float* Whh_f, const float* b_f,
    const float* Wih_b, const float* Whh_b, const float* b_b, float* char_vec){
  int dir = blockIdx.y;
  const float* Wih = dir ? Wih_b : Wih_f;
  const float* Whh = dir ? Whh_b : Whh_f;
  const float* bb  = dir ? b_b  : b_f;
  int n0 = blockIdx.x * 8;
  __shared__ float xs[8][64];
  __shared__ float hs[8][152];
  __shared__ int   cidx[8];
  int j = threadIdx.x;
  float c[8], hn[8];
  if (j < 150){
#pragma unroll
    for (int g = 0; g < 8; g++){ c[g] = 0.f; hs[g][j] = 0.f; }
  }
  float bi=0.f, bff=0.f, bg=0.f, bo=0.f;
  if (j < 150){ bi=bb[j]; bff=bb[150+j]; bg=bb[300+j]; bo=bb[450+j]; }
  for (int step = 0; step < 8; step++){
    int t = dir ? 7 - step : step;
    if (j < 8) cidx[j] = char_idxs[(n0 + j)*8 + t];
    __syncthreads();
    for (int i = j; i < 512; i += 160){ int g = i >> 6, kk = i & 63; xs[g][kk] = char_emb[cidx[g]*64 + kk]; }
    __syncthreads();
    if (j < 150){
      float ai[8], af[8], ag[8], ao[8];
#pragma unroll
      for (int g = 0; g < 8; g++){ ai[g]=bi; af[g]=bff; ag[g]=bg; ao[g]=bo; }
      for (int k = 0; k < 64; k++){
        float wi=Wih[k*600+j], wf=Wih[k*600+150+j], wg=Wih[k*600+300+j], wo=Wih[k*600+450+j];
#pragma unroll
        for (int g = 0; g < 8; g++){ float xv = xs[g][k]; ai[g]+=xv*wi; af[g]+=xv*wf; ag[g]+=xv*wg; ao[g]+=xv*wo; }
      }
      for (int k = 0; k < 150; k++){
        float wi=Whh[k*600+j], wf=Whh[k*600+150+j], wg=Whh[k*600+300+j], wo=Whh[k*600+450+j];
#pragma unroll
        for (int g = 0; g < 8; g++){ float hv = hs[g][k]; ai[g]+=hv*wi; af[g]+=hv*wf; ag[g]+=hv*wg; ao[g]+=hv*wo; }
      }
#pragma unroll
      for (int g = 0; g < 8; g++){
        float ii=sigm(ai[g]), ff=sigm(af[g]), gg=tanhf(ag[g]), oo=sigm(ao[g]);
        c[g] = ff*c[g] + ii*gg;
        hn[g] = oo*tanhf(c[g]);
      }
    }
    __syncthreads();
    if (j < 150){
#pragma unroll
      for (int g = 0; g < 8; g++) hs[g][j] = hn[g];
    }
    __syncthreads();
  }
  if (j < 150)
    for (int g = 0; g < 8; g++)
      char_vec[(long long)(n0 + g)*300 + dir*150 + j] = hs[g][j];
}

// ---------------------------------------------------- embeddings + LN0
__global__ __launch_bounds__(256) void k_embed_ln0(const int* word_idxs, const float* ext_emb,
    const float* word_emb, const float* char_vec, const float* pos_emb,
    const float* g, const float* be, float* x, bf16_t* xbf){
  int n = blockIdx.x, s = n & 255;
  int w = word_idxs[n];
  int m = (w != 0);
  int wid = (w >= VTRAIN) ? 1 : w;
  int pos = (s + 1) * m;
  int t = threadIdx.x;
  float v0 = ext_emb[(long long)w*300 + t] + word_emb[(long long)wid*300 + t] + char_vec[(long long)n*300 + t];
  float v1;
  {
    int c = t + 256;
    if (c < 300) v1 = ext_emb[(long long)w*300 + c] + word_emb[(long long)wid*300 + c] + char_vec[(long long)n*300 + c];
    else         v1 = pos_emb[(long long)pos*212 + (c - 300)];
  }
  float mean = blk_sum(v0 + v1) * (1.f/512.f);
  float d0 = v0 - mean, d1 = v1 - mean;
  float var = blk_sum(d0*d0 + d1*d1) * (1.f/512.f);
  float rstd = rsqrtf(var + 1e-5f);
  float o0 = d0*rstd*g[t] + be[t];
  float o1 = d1*rstd*g[t+256] + be[t+256];
  long long bse = (long long)n*512;
  x[bse + t] = o0;        xbf[bse + t] = f2bf(o0);
  x[bse + t + 256] = o1;  xbf[bse + t + 256] = f2bf(o1);
}

// ---------------------------------------------------- residual + LN (+bf16)
__global__ __launch_bounds__(256) void k_res_ln(float* x, const float* y,
    const float* g, const float* be, bf16_t* xbf){
  int n = blockIdx.x, t = threadIdx.x;
  long long bse = (long long)n*512;
  float v0 = x[bse + t] + y[bse + t];
  float v1 = x[bse + t + 256] + y[bse + t + 256];
  float mean = blk_sum(v0 + v1) * (1.f/512.f);
  float d0 = v0 - mean, d1 = v1 - mean;
  float var = blk_sum(d0*d0 + d1*d1) * (1.f/512.f);
  float rstd = rsqrtf(var + 1e-5f);
  float o0 = d0*rstd*g[t] + be[t];
  float o1 = d1*rstd*g[t+256] + be[t+256];
  x[bse + t] = o0;        xbf[bse + t] = f2bf(o0);
  x[bse + t + 256] = o1;  xbf[bse + t + 256] = f2bf(o1);
}

// ---------------------------------------------------- masked softmax -> bf16
__global__ __launch_bounds__(256) void k_softmax(const float* sc, bf16_t* pr, const int* wi){
  int wave = threadIdx.x >> 5, lane = threadIdx.x & 31;
  long long row = (long long)blockIdx.x*8 + wave;   // (b,h,q)
  int b = (int)(row >> 11);
  const float* s = sc + row*256;
  float v[8]; float mx = -3.4e38f;
#pragma unroll
  for (int ii = 0; ii < 8; ii++){
    int k = ii*32 + lane;
    float t = s[k]*0.125f + ((wi[b*256 + k] != 0) ? 0.f : -1.0e9f);
    v[ii] = t; mx = fmaxf(mx, t);
  }
  for (int m = 16; m; m >>= 1) mx = fmaxf(mx, __shfl_xor(mx, m, 32));
  float sum = 0.f;
#pragma unroll
  for (int ii = 0; ii < 8; ii++){ v[ii] = expf(v[ii] - mx); sum += v[ii]; }
  for (int m = 16; m; m >>= 1) sum += __shfl_xor(sum, m, 32);
  float inv = 1.f / sum;
#pragma unroll
  for (int ii = 0; ii < 8; ii++){ int k = ii*32 + lane; pr[row*256 + k] = f2bf(v[ii]*inv); }
}

// ---------------------------------------------------- span differences
DEV int tri_off(int p){ return p*254 - (p*(p-1))/2; }

__global__ __launch_bounds__(256) void k_span(const float* x, const int* word_idxs, float* out){
  int blk = blockIdx.x;
  int b = blk / NPAIR;
  int t = blk - b*NPAIR;
  int p = (int)((509.0 - sqrt(509.0*509.0 - 8.0*(double)t)) * 0.5);
  if (p < 0) p = 0; if (p > 253) p = 253;
  while (p > 0 && tri_off(p) > t) --p;
  while (tri_off(p + 1) <= t) ++p;
  int q = t - tri_off(p) + p + 1;
  int wq = word_idxs[b*256 + q];
  float vf = (wq != 0 && wq != 2) ? 1.f : 0.f;
  int i = threadIdx.x;
  const float* xb = x + (long long)b*256*512;
  float fwd = (xb[(long long)q*512 + 2*i]     - xb[(long long)p*512 + 2*i])     * vf;
  float bwd = (xb[(long long)(p+1)*512 + 2*i + 1] - xb[(long long)(q+1)*512 + 2*i + 1]) * vf;
  long long base = (long long)blk * 512;
  out[base + i] = fwd;
  out[base + 256 + i] = bwd;
}

__global__ void k_senlens(const int* wi, float* out){
  int b = threadIdx.x;
  if (b < 8){
    int c = 0;
    for (int s = 0; s < 256; s++) c += (wi[b*256 + s] != 0);
    out[b] = (float)(c - 2);
  }
}

// ======================================================================
extern "C" void kernel_launch(void* const* d_in, const int* in_sizes, int n_in,
                              void* d_out, int out_size, void* d_ws, size_t ws_size,
                              hipStream_t stream){
  (void)in_sizes; (void)n_in; (void)out_size; (void)ws_size;
  const int*   word_idxs = (const int*)d_in[0];
  const int*   char_idxs = (const int*)d_in[1];
  const float* ext_emb = (const float*)d_in[2];
  const float* word_emb= (const float*)d_in[3];
  const float* char_emb= (const float*)d_in[4];
  const float* Wih_f = (const float*)d_in[5];
  const float* Whh_f = (const float*)d_in[6];
  const float* b_f   = (const float*)d_in[7];
  const float* Wih_b = (const float*)d_in[8];
  const float* Whh_b = (const float*)d_in[9];
  const float* b_b   = (const float*)d_in[10];
  const float* pos_emb = (const float*)d_in[11];
  const float* ln0_g = (const float*)d_in[12];
  const float* ln0_b = (const float*)d_in[13];
  const float* Wq = (const float*)d_in[14];
  const float* Wk = (const float*)d_in[15];
  const float* Wv = (const float*)d_in[16];
  const float* Wo = (const float*)d_in[17];
  const float* ln1_g = (const float*)d_in[18];
  const float* ln1_b = (const float*)d_in[19];
  const float* W1 = (const float*)d_in[20];
  const float* b1 = (const float*)d_in[21];
  const float* W2 = (const float*)d_in[22];
  const float* b2 = (const float*)d_in[23];
  const float* ln2_g = (const float*)d_in[24];
  const float* ln2_b = (const float*)d_in[25];

  size_t off = 0;
  char* base = (char*)d_ws;
  auto alloc = [&](size_t bytes)->void*{
    void* p = base + off; off = (off + bytes + 255) & ~(size_t)255; return p; };

  const size_t SQ = 512*512, SFF = 512*2048;
  const size_t PERL = 4*SQ + 2*SFF;
  bf16_t* WBT     = (bf16_t*)alloc(PERL * LLAY * sizeof(bf16_t));
  float*  char_vec= (float*) alloc((size_t)2048*300*4);
  float*  x       = (float*) alloc((size_t)2048*512*4);
  bf16_t* xbf     = (bf16_t*)alloc((size_t)2048*512*2);
  bf16_t* qb      = (bf16_t*)alloc((size_t)2048*512*2);
  bf16_t* kb      = (bf16_t*)alloc((size_t)2048*512*2);
  bf16_t* vt      = (bf16_t*)alloc((size_t)2048*512*2);
  float*  scores  = (float*) alloc((size_t)64*256*256*4);
  bf16_t* probs   = (bf16_t*)alloc((size_t)64*256*256*2);
  bf16_t* ob      = (bf16_t*)alloc((size_t)2048*512*2);
  float*  tmp     = (float*) alloc((size_t)2048*512*4);
  bf16_t* hbuf    = (bf16_t*)alloc((size_t)2048*2048*2);

  // one-time weight transpose + f32->bf16 (all L2-resident afterwards)
  for (int l = 0; l < LLAY; l++){
    bf16_t* Wl = WBT + (size_t)l*PERL;
    k_transpose<<<(SQ +255)/256,256,0,stream>>>(Wq + l*SQ,  Wl + 0*SQ, 512, 512);
    k_transpose<<<(SQ +255)/256,256,0,stream>>>(Wk + l*SQ,  Wl + 1*SQ, 512, 512);
    k_transpose<<<(SQ +255)/256,256,0,stream>>>(Wv + l*SQ,  Wl + 2*SQ, 512, 512);
    k_transpose<<<(SQ +255)/256,256,0,stream>>>(Wo + l*SQ,  Wl + 3*SQ, 512, 512);
    k_transpose<<<(SFF+255)/256,256,0,stream>>>(W1 + l*SFF, Wl + 4*SQ, 512, 2048);
    k_transpose<<<(SFF+255)/256,256,0,stream>>>(W2 + l*SFF, Wl + 4*SQ + SFF, 2048, 512);
  }

  k_lstm<<<dim3(256,2),160,0,stream>>>(char_idxs, char_emb,
      Wih_f, Whh_f, b_f, Wih_b, Whh_b, b_b, char_vec);
  k_embed_ln0<<<2048,256,0,stream>>>(word_idxs, ext_emb, word_emb,
      char_vec, pos_emb, ln0_g, ln0_b, x, xbf);

  auto gemm = [&](const bf16_t* A, long long Aso, long long Asi, int lda,
                  const bf16_t* Bt, long long Bso, long long Bsi, int ldb,
                  float* Cf, bf16_t* Cb, long long Cso, long long Csi, int ldc,
                  int M, int N, int K, int Z, int Hdiv,
                  const float* bias, int relu, int mode){
    GemmP p{A,Aso,Asi,lda, Bt,Bso,Bsi,ldb, Cf,Cb,Cso,Csi,ldc, K,Hdiv,bias,relu,mode};
    dim3 grid(N/64, M/128, Z);
    k_gemm<<<grid,128,0,stream>>>(p);
  };

  const long long SD = 256LL*512, SSQ = 256LL*256;
  for (int l = 0; l < LLAY; l++){
    bf16_t* Wl = WBT + (size_t)l*PERL;
    bf16_t *WqT=Wl, *WkT=Wl+SQ, *WvT=Wl+2*SQ, *WoT=Wl+3*SQ, *W1T=Wl+4*SQ, *W2T=Wl+4*SQ+SFF;

    // Q, K (bf16), V (bf16 transposed per head: Vt[b,h,d,s])
    gemm(xbf,0,0,512,  WqT,0,0,512,  nullptr,qb, 0,0,512,  2048,512,512, 1,1, nullptr,0,1);
    gemm(xbf,0,0,512,  WkT,0,0,512,  nullptr,kb, 0,0,512,  2048,512,512, 1,1, nullptr,0,1);
    gemm(xbf,0,0,512,  WvT,0,0,512,  nullptr,vt, 0,0,512,  2048,512,512, 1,1, nullptr,0,2);

    // scores[b,h,q,k] = Q . K^T  (K matrix row-major *is* Bt of K^T)
    gemm(qb,SD,64,512,  kb,SD,64,512,  scores,nullptr, 8*SSQ,SSQ,256,  256,256,64, 64,8, nullptr,0,0);
    k_softmax<<<2048,256,0,stream>>>(scores, probs, word_idxs);

    // O[b,q,h*64+d] = probs @ V   (Bt = Vt)
    gemm(probs,8*SSQ,SSQ,256,  vt,8*16384LL,16384,256,  nullptr,ob, SD,64,512,  256,64,256, 64,8, nullptr,0,1);

    // output projection, residual + LN1
    gemm(ob,0,0,512,  WoT,0,0,512,  tmp,nullptr,0,0,512,  2048,512,512, 1,1, nullptr,0,0);
    k_res_ln<<<2048,256,0,stream>>>(x, tmp, ln1_g + l*512, ln1_b + l*512, xbf);

    // FFN: relu(x@W1+b1) -> bf16, then @W2+b2 -> f32, residual + LN2
    gemm(xbf,0,0,512,   W1T,0,0,512,   nullptr,hbuf, 0,0,2048,  2048,2048,512, 1,1, b1 + l*2048, 1, 1);
    gemm(hbuf,0,0,2048, W2T,0,0,2048,  tmp,nullptr,  0,0,512,   2048,512,2048, 1,1, b2 + l*512,  0, 0);
    k_res_ln<<<2048,256,0,stream>>>(x, tmp, ln2_g + l*512, ln2_b + l*512, xbf);
  }

  float* out = (float*)d_out;
  k_span<<<BB*NPAIR,256,0,stream>>>(x, word_idxs, out);
  k_senlens<<<1,32,0,stream>>>(word_idxs, out + (long long)BB*NPAIR*512);
}
